// DummySAE_6382321402556
// MI455X (gfx1250) — matrix operand order repaired and checked by reading, not compiled, not run on hardware
//
#include <hip/hip_runtime.h>
#include <hip/hip_bf16.h>

typedef __attribute__((ext_vector_type(16))) _Float16 v16h;
typedef __attribute__((ext_vector_type(8)))  float    v8f;

#define BB   2
#define SS   2048
#define KK   32
#define EE   2048          // N_EMBD
#define NTOK (BB * SS)     // 4096 tokens

// One block (256 threads = 8 waves, wave32) per token.
// Each wave computes 16 chunks of 16 embedding columns via one
// v_wmma_f32_16x16x32_f16 per chunk:
//   D[16x16] = A[16x32] * B[32x16] + 0
//   A[m,k] = feat_values[tok,k]          (broadcast across m)
//   B[k,n] = W_dec[idx[tok,k], e0+n]     (the gather)
// All 16 rows of D are identical. Lane L holds D[M = (L<16?0:8), n=L&15]
// in element 0 -- same value and same output address for lane j and
// lane j+16, so ALL 32 lanes store unconditionally (duplicate-equal
// writes). No divergence anywhere => EXEC all-ones at every WMMA and
// no sinking of the bias loads into a branch.
__global__ __launch_bounds__(256) void sae_decode_wmma(
    const float* __restrict__ feat_values,   // [NTOK, 32] f32
    const int*   __restrict__ feat_indices,  // [NTOK, 32] i32
    const float* __restrict__ W_dec,         // [32768, 2048] f32
    const float* __restrict__ b_dec,         // [2048] f32
    float*       __restrict__ out)           // [NTOK, 2048] f32
{
    __shared__ int   sIdx[KK];
    __shared__ float sVal[KK];

    const int tok = blockIdx.x;
    const int t   = threadIdx.x;
    if (t < KK)            sIdx[t]      = feat_indices[(size_t)tok * KK + t];
    else if (t < 2 * KK)   sVal[t - KK] = feat_values [(size_t)tok * KK + (t - KK)];
    __syncthreads();

    const int wave   = t >> 5;        // 0..7
    const int lane   = t & 31;
    const int half   = lane >> 4;     // 0 -> owns K 0..15, 1 -> owns K 16..31
    const int lane15 = lane & 15;

    // Wave's first column (in floats) within the token's 2048-wide row.
    const unsigned col0 = (unsigned)(wave * 16 + lane15);

    // 16 per-lane row pointers, computed once; the 16 chunk steps
    // (+512B each) fold into each load's 24-bit immediate offset.
    const float* rowPtr[16];
#pragma unroll
    for (int j = 0; j < 16; ++j)
        rowPtr[j] = W_dec + (size_t)((unsigned)sIdx[half * 16 + j] * (unsigned)EE + col0);

    // A matrix (16x32 f16), v broadcast down M. Contiguous K order
    // (matches B): packed slot p holds K = half*16 + p.
    v16h amat;
#pragma unroll
    for (int p = 0; p < 16; ++p) {
        int k = half * 16 + p;
        amat[p] = (_Float16)sVal[k];
    }

    // Preload the wave's 16 per-chunk bias values up front (imm-offset
    // clause); used unconditionally so they stay hoisted.
    float bias[16];
#pragma unroll
    for (int it = 0; it < 16; ++it)
        bias[it] = b_dec[col0 + (unsigned)(it * 128)];

    float* outTok = out + (size_t)tok * EE + col0;

    const v8f czero = {};   // loop-invariant C operand (inline 0)

#pragma unroll
    for (int it = 0; it < 16; ++it) {
        // Gather B (32x16): VGPR r packs rows K = half*16 + {2r, 2r+1}
        // at column col0 + it*128. Each b32 load = two contiguous 64B
        // segments (one per half-wave row).
        v16h bmat;
#pragma unroll
        for (int r = 0; r < 8; ++r) {
            float f0 = rowPtr[2 * r]    [it * 128];
            float f1 = rowPtr[2 * r + 1][it * 128];
            bmat[2 * r]     = (_Float16)f0;
            bmat[2 * r + 1] = (_Float16)f1;
        }

        v8f d = __builtin_amdgcn_wmma_f32_16x16x32_f16(
            /*neg_a=*/false, amat, /*neg_b=*/false, bmat,
            /*c_mod=*/(short)0, czero, /*reuse_a=*/false, /*reuse_b=*/false);

        // Element 0: lane<16 -> D[0, lane]; lane>=16 -> D[8, lane-16].
        // Identical values, identical addresses pairwise: store from all
        // lanes, branch-free.
        outTok[it * 128] = d[0] + bias[it];
    }
}

extern "C" void kernel_launch(void* const* d_in, const int* in_sizes, int n_in,
                              void* d_out, int out_size, void* d_ws, size_t ws_size,
                              hipStream_t stream) {
    const float* feat_values  = (const float*)d_in[0];
    const int*   feat_indices = (const int*)  d_in[1];
    const float* W_dec        = (const float*)d_in[2];
    const float* b_dec        = (const float*)d_in[3];
    float*       out          = (float*)d_out;

    dim3 grid(NTOK);
    dim3 block(256);
    hipLaunchKernelGGL(sae_decode_wmma, grid, block, 0, stream,
                       feat_values, feat_indices, W_dec, b_dec, out);
}
